// MarkdownModel_72138270703813
// MI455X (gfx1250) — matrix operand order repaired and checked by reading, not compiled, not run on hardware
//
#include <hip/hip_runtime.h>
#include <math.h>

// ---------------- model dims ----------------
constexpr int Bz   = 2;
constexpr int Sq   = 4096;
constexpr int Hh   = 12;
constexpr int Dd   = 64;
constexpr int Ll   = 12;
constexpr int FFd  = 3072;
constexpr int HIDc = 768;
constexpr int BLKc = 256;   // query block
constexpr int WINc = 256;   // one-sided window
constexpr int Mrows = Bz * Sq;          // 8192

// ---------------- WMMA / vector types ----------------
typedef __attribute__((ext_vector_type(16))) __bf16 v16bf;
typedef __attribute__((ext_vector_type(8)))  float  v8f;
typedef __attribute__((ext_vector_type(4)))  unsigned int v4u;
typedef __attribute__((ext_vector_type(8)))  int    v8i;
typedef __attribute__((ext_vector_type(4)))  int    v4i;

union FragBf {
    v16bf  v;
    uint4  q[2];
    __bf16 e[16];
};
union U8 {
    uint4  u;
    __bf16 e[8];
};

__device__ inline v8f zero8() {
    v8f z;
    #pragma unroll
    for (int i = 0; i < 8; ++i) z[i] = 0.0f;
    return z;
}

__device__ inline v8f wmma_bf16(const FragBf& a, const FragBf& b, v8f c) {
    return __builtin_amdgcn_wmma_f32_16x16x32_bf16(
        false, a.v, false, b.v, (short)0, c, false, false);
}

__device__ inline float gelu_exact(float x) {
    return 0.5f * x * (1.0f + erff(x * 0.70710678118654752f));
}

// =====================================================================
// TDM: async 2-D bf16 tile load global -> LDS with per-row padding.
// Tile: tile_y rows x tile_x bf16 (contiguous), row stride = stride elems.
// LDS rows are padded: 16 DWORDs data + 4 DWORDs pad (pad_interval=3,
// pad_amount=3) -> LDS row pitch = (tile_x + 8) bf16 when tile_x == 32.
// Tracked on TENSORcnt.  (6-arg builtin: g0,g1,g2,g3,g4,cpol)
// =====================================================================
__device__ inline void tdm_load_2d_bf16(const __bf16* gaddr, unsigned lds_off,
                                        int tile_x, int tile_y, int stride_elems)
{
    const unsigned long long ga = (unsigned long long)(const void*)gaddr;
    v4u g0;
    g0[0] = 1u;                                            // count=1, user mode
    g0[1] = lds_off;                                       // LDS byte address
    g0[2] = (unsigned)(ga & 0xffffffffu);                  // global_addr[31:0]
    g0[3] = (unsigned)((ga >> 32) & 0x1ffffffu) | (2u << 30); // addr[56:32] | type=2

    const unsigned td0 = (unsigned)tile_x;                 // tensor_dim0 (= tile)
    const unsigned td1 = (unsigned)tile_y;                 // tensor_dim1 (= tile)
    const unsigned st0 = (unsigned)stride_elems;           // tensor_dim0_stride

    v8i g1;
    g1[0] = (int)((1u << 16)      // data_size = 2 bytes (bf16)
                | (1u << 20)      // pad_enable
                | (3u << 22)      // pad_interval: 16 DWORDs
                | (3u << 25));    // pad_amount: 4 DWORDs
    g1[1] = (int)((td0 & 0xffffu) << 16);                          // tensor_dim0 lo
    g1[2] = (int)(((td0 >> 16) & 0xffffu) | ((td1 & 0xffffu) << 16));
    g1[3] = (int)(((td1 >> 16) & 0xffffu) | ((unsigned)tile_x << 16)); // tile_dim0
    g1[4] = (int)((unsigned)tile_y & 0xffffu);                     // tile_dim1
    g1[5] = (int)st0;                                              // dim0_stride lo
    g1[6] = 0;                                                     // stride hi
    g1[7] = 0;

    v4i z4;
    z4[0] = 0; z4[1] = 0; z4[2] = 0; z4[3] = 0;
    v8i z8;
    #pragma unroll
    for (int i = 0; i < 8; ++i) z8[i] = 0;
    __builtin_amdgcn_tensor_load_to_lds(g0, g1, z4, z4, z8, 0);
}

// =====================================================================
// Weight convert + transpose: in fp32 [K,N] (per layer) -> out bf16 [N,K].
// grid = (N/64, K/64, L), block = 256
// =====================================================================
__global__ void convt_kernel(const float* __restrict__ in, __bf16* __restrict__ out,
                             int K, int N)
{
    __shared__ __align__(16) __bf16 sT[64][72];
    const size_t mat = (size_t)blockIdx.z * K * N;
    const float* src = in + mat;
    __bf16* dst = out + mat;
    const int n0 = blockIdx.x * 64, k0 = blockIdx.y * 64;
    const int t = threadIdx.x;
    const int kr = t >> 2, cs = (t & 3) * 16;
    const float* p = src + (size_t)(k0 + kr) * N + n0 + cs;
    #pragma unroll
    for (int i = 0; i < 16; i += 4) {
        const float4 f = *(const float4*)(p + i);
        sT[kr][cs + i + 0] = (__bf16)f.x;
        sT[kr][cs + i + 1] = (__bf16)f.y;
        sT[kr][cs + i + 2] = (__bf16)f.z;
        sT[kr][cs + i + 3] = (__bf16)f.w;
    }
    __syncthreads();
    const int nr = t >> 2, ks2 = (t & 3) * 16;
    __bf16* q = dst + (size_t)(n0 + nr) * K + k0 + ks2;
    #pragma unroll
    for (int i = 0; i < 16; ++i) q[i] = sT[ks2 + i][nr];
}

// =====================================================================
// Embedding gather + LayerNorm -> bf16 h.  grid = Mrows, block = 256
// =====================================================================
__global__ void embed_ln_kernel(const int* __restrict__ ids,
                                const float* __restrict__ wemb,
                                const float* __restrict__ pemb,
                                const float* __restrict__ temb,
                                const float* __restrict__ g,
                                const float* __restrict__ b,
                                __bf16* __restrict__ h)
{
    const int tok = blockIdx.x;
    const int s   = tok % Sq;
    const int id  = ids[tok];
    const int tid = threadIdx.x;

    float v[3];
    #pragma unroll
    for (int r = 0; r < 3; ++r) {
        const int c = tid + r * 256;
        v[r] = wemb[(size_t)id * HIDc + c] + pemb[(size_t)s * HIDc + c] + temb[c];
    }
    float s1 = v[0] + v[1] + v[2];
    float s2 = v[0]*v[0] + v[1]*v[1] + v[2]*v[2];
    #pragma unroll
    for (int o = 16; o; o >>= 1) { s1 += __shfl_xor(s1, o); s2 += __shfl_xor(s2, o); }
    __shared__ float rs1[8], rs2[8];
    if ((tid & 31) == 0) { rs1[tid >> 5] = s1; rs2[tid >> 5] = s2; }
    __syncthreads();
    s1 = 0.0f; s2 = 0.0f;
    #pragma unroll
    for (int i = 0; i < 8; ++i) { s1 += rs1[i]; s2 += rs2[i]; }
    const float mean = s1 * (1.0f / HIDc);
    const float var  = s2 * (1.0f / HIDc) - mean * mean;
    const float inv  = rsqrtf(var + 1e-5f);
    __bf16* out = h + (size_t)tok * HIDc;
    #pragma unroll
    for (int r = 0; r < 3; ++r) {
        const int c = tid + r * 256;
        out[c] = (__bf16)((v[r] - mean) * inv * g[c] + b[c]);
    }
}

// =====================================================================
// In-place LayerNorm on bf16 rows. grid = Mrows, block = 256
// =====================================================================
__global__ void ln_kernel(__bf16* __restrict__ x,
                          const float* __restrict__ g,
                          const float* __restrict__ b)
{
    const int tid = threadIdx.x;
    __bf16* row = x + (size_t)blockIdx.x * HIDc;
    float v[3];
    #pragma unroll
    for (int r = 0; r < 3; ++r) v[r] = (float)row[tid + r * 256];
    float s1 = v[0] + v[1] + v[2];
    float s2 = v[0]*v[0] + v[1]*v[1] + v[2]*v[2];
    #pragma unroll
    for (int o = 16; o; o >>= 1) { s1 += __shfl_xor(s1, o); s2 += __shfl_xor(s2, o); }
    __shared__ float rs1[8], rs2[8];
    if ((tid & 31) == 0) { rs1[tid >> 5] = s1; rs2[tid >> 5] = s2; }
    __syncthreads();
    s1 = 0.0f; s2 = 0.0f;
    #pragma unroll
    for (int i = 0; i < 8; ++i) { s1 += rs1[i]; s2 += rs2[i]; }
    const float mean = s1 * (1.0f / HIDc);
    const float var  = s2 * (1.0f / HIDc) - mean * mean;
    const float inv  = rsqrtf(var + 1e-5f);
    #pragma unroll
    for (int r = 0; r < 3; ++r) {
        const int c = tid + r * 256;
        row[c] = (__bf16)((v[r] - mean) * inv * g[c] + b[c]);
    }
}

// =====================================================================
// WMMA GEMM with TDM double-buffered staging.
//   C[M,N] = epi(A[M,K] @ WtT[N,K]^T + bias [+ Res]), all act/weights bf16.
//   Block tile 128x128, 8 waves, wave tile 32x64 (2x4 WMMA accumulators).
//   Both operands are [rows][K]-major bf16; a 128x32 tile of each is DMA'd
//   by the Tensor Data Mover into padded LDS (row pitch 40 bf16) while the
//   previous tile is consumed (TENSORcnt-tracked double buffering).
// EPI: 0 = bias, 1 = bias+GELU, 2 = bias+residual
// =====================================================================
template<int EPI>
__global__ __launch_bounds__(256)
void gemm_kernel(const __bf16* __restrict__ A, const __bf16* __restrict__ Wt,
                 const float* __restrict__ bias, const __bf16* __restrict__ Res,
                 __bf16* __restrict__ C, int M, int N, int K)
{
    // [buf0 A][buf1 A][buf0 B][buf1 B], each 128 rows x 40 bf16
    __shared__ __align__(16) __bf16 smem[4 * 128 * 40];

    const int tid  = threadIdx.x;
    const int lane = tid & 31, wid = tid >> 5;
    const int hi   = lane >> 4, ln = lane & 15;
    const int wr   = wid >> 1,  wc = wid & 1;      // 4x2 wave grid
    const int rowBase = blockIdx.y * 128;
    const int colBase = blockIdx.x * 128;

    const unsigned ldsBase = (unsigned)(unsigned long long)(void*)smem;
    const unsigned bufBytes = 128u * 40u * 2u;

    v8f acc[2][4];
    #pragma unroll
    for (int mi = 0; mi < 2; ++mi)
        #pragma unroll
        for (int ni = 0; ni < 4; ++ni) acc[mi][ni] = zero8();

    const __bf16* Abase = A  + (size_t)rowBase * K;
    const __bf16* Bbase = Wt + (size_t)colBase * K;

    if (wid == 0) {
        tdm_load_2d_bf16(Abase, ldsBase,                32, 128, K);
        tdm_load_2d_bf16(Bbase, ldsBase + 2 * bufBytes, 32, 128, K);
    }

    const int nk = K / 32;
    for (int ik = 0; ik < nk; ++ik) {
        const int buf = ik & 1;
        if (wid == 0) {
            if (ik + 1 < nk) {
                tdm_load_2d_bf16(Abase + (ik + 1) * 32,
                                 ldsBase + (unsigned)(buf ^ 1) * bufBytes, 32, 128, K);
                tdm_load_2d_bf16(Bbase + (ik + 1) * 32,
                                 ldsBase + 2 * bufBytes + (unsigned)(buf ^ 1) * bufBytes,
                                 32, 128, K);
                __builtin_amdgcn_s_wait_tensorcnt(2);   // tile ik complete
            } else {
                __builtin_amdgcn_s_wait_tensorcnt(0);
            }
        }
        __syncthreads();                                // LDS tile published

        const __bf16* aTile = smem + buf * (128 * 40);
        const __bf16* bTile = smem + (2 + buf) * (128 * 40);

        FragBf af[2], bfb[4];
        #pragma unroll
        for (int mi = 0; mi < 2; ++mi) {
            const int r = wr * 32 + mi * 16 + ln;
            af[mi].q[0] = *(const uint4*)(aTile + r * 40 + hi * 8);
            af[mi].q[1] = *(const uint4*)(aTile + r * 40 + hi * 8 + 16);
        }
        #pragma unroll
        for (int ni = 0; ni < 4; ++ni) {
            const int cN = wc * 64 + ni * 16 + ln;
            bfb[ni].q[0] = *(const uint4*)(bTile + cN * 40 + hi * 8);
            bfb[ni].q[1] = *(const uint4*)(bTile + cN * 40 + hi * 8 + 16);
        }
        #pragma unroll
        for (int mi = 0; mi < 2; ++mi)
            #pragma unroll
            for (int ni = 0; ni < 4; ++ni)
                acc[mi][ni] = wmma_bf16(af[mi], bfb[ni], acc[mi][ni]);
        __syncthreads();                                // done reading buf
    }

    // ---- epilogue ----
    #pragma unroll
    for (int mi = 0; mi < 2; ++mi)
        #pragma unroll
        for (int ni = 0; ni < 4; ++ni) {
            const int cN = colBase + wc * 64 + ni * 16 + ln;
            #pragma unroll
            for (int j = 0; j < 8; ++j) {
                const int r = rowBase + wr * 32 + mi * 16 + hi * 8 + j;
                float val = acc[mi][ni][j] + bias[cN];
                if constexpr (EPI == 1) val = gelu_exact(val);
                if constexpr (EPI == 2) val += (float)Res[(size_t)r * N + cN];
                C[(size_t)r * N + cN] = (__bf16)val;
            }
        }
}

// =====================================================================
// Sliding-window flash attention (band |i-j| <= 256), bf16 in/out.
// grid = (Sq/BLKc, Hh, Bz), block = 256 (8 waves x 32 query rows).
// Q fragments load straight from global (bf16 A-frag layout is two
// contiguous 16B runs per lane); K/V chunks staged to LDS as uint4.
// =====================================================================
__global__ __launch_bounds__(256)
void attn_kernel(const __bf16* __restrict__ Q, const __bf16* __restrict__ Kg,
                 const __bf16* __restrict__ Vg, __bf16* __restrict__ Ctx)
{
    const int nb = blockIdx.x, hh = blockIdx.y, bb = blockIdx.z;
    const int tid = threadIdx.x;
    const int lane = tid & 31, wid = tid >> 5;
    const int hi = lane >> 4, ln = lane & 15;

    const size_t baseBH = (size_t)bb * Sq * HIDc + (size_t)hh * Dd;
    const __bf16* Qp = Q  + baseBH;
    const __bf16* Kp = Kg + baseBH;
    const __bf16* Vp = Vg + baseBH;
    __bf16*       Cp = Ctx + baseBH;

    __shared__ __align__(16) __bf16 sK [32][72];       // [key][d]
    __shared__ __align__(16) __bf16 sVT[64][40];       // [d][key]
    __shared__ __align__(16) __bf16 sP [8][32][40];    // per-wave P tile [q][key]

    const int q0 = nb * BLKc + wid * 32;

    // ---- Q fragments: direct global b128 loads ----
    FragBf qf[2][2];
    #pragma unroll
    for (int mi = 0; mi < 2; ++mi) {
        const __bf16* qrow = Qp + (size_t)(q0 + mi * 16 + ln) * HIDc;
        #pragma unroll
        for (int ks = 0; ks < 2; ++ks) {
            qf[mi][ks].q[0] = *(const uint4*)(qrow + ks * 32 + hi * 8);
            qf[mi][ks].q[1] = *(const uint4*)(qrow + ks * 32 + hi * 8 + 16);
        }
    }

    v8f oacc[2][4];
    #pragma unroll
    for (int mi = 0; mi < 2; ++mi)
        #pragma unroll
        for (int ni = 0; ni < 4; ++ni) oacc[mi][ni] = zero8();
    float mrow[16], lrow[16];
    #pragma unroll
    for (int i = 0; i < 16; ++i) { mrow[i] = -1e30f; lrow[i] = 0.0f; }

    const int kstart = nb * BLKc - BLKc;

    for (int c = 0; c < 24; ++c) {          // 24 * 32 = 768 banded keys
        const int kc0 = kstart + c * 32;

        // ---- stage K chunk and V^T chunk (bf16, uint4 granularity) ----
        {
            const int key = tid >> 3;
            const int d0  = (tid & 7) * 8;
            const int kabs = kc0 + key;
            U8 kv, vv;
            if (kabs >= 0 && kabs < Sq) {
                kv.u = *(const uint4*)(Kp + (size_t)kabs * HIDc + d0);
                vv.u = *(const uint4*)(Vp + (size_t)kabs * HIDc + d0);
            } else {
                kv.u.x = kv.u.y = kv.u.z = kv.u.w = 0u;
                vv.u = kv.u;
            }
            *(uint4*)&sK[key][d0] = kv.u;
            #pragma unroll
            for (int i = 0; i < 8; ++i) sVT[d0 + i][key] = vv.e[i];
        }
        __syncthreads();

        // ---- scores: S = Q K^T  (32 queries x 32 keys per wave) ----
        v8f sacc[2][2];
        #pragma unroll
        for (int mi = 0; mi < 2; ++mi)
            #pragma unroll
            for (int ci = 0; ci < 2; ++ci) sacc[mi][ci] = zero8();
        #pragma unroll
        for (int ks = 0; ks < 2; ++ks) {
            FragBf kf[2];
            #pragma unroll
            for (int ci = 0; ci < 2; ++ci) {
                const int keyloc = ci * 16 + ln;
                kf[ci].q[0] = *(const uint4*)&sK[keyloc][ks * 32 + hi * 8];
                kf[ci].q[1] = *(const uint4*)&sK[keyloc][ks * 32 + hi * 8 + 16];
            }
            #pragma unroll
            for (int mi = 0; mi < 2; ++mi)
                #pragma unroll
                for (int ci = 0; ci < 2; ++ci)
                    sacc[mi][ci] = wmma_bf16(qf[mi][ks], kf[ci], sacc[mi][ci]);
        }

        // ---- masking + online softmax (scale 1/sqrt(64) applied here) ----
        #pragma unroll
        for (int mi = 0; mi < 2; ++mi) {
            #pragma unroll
            for (int j = 0; j < 8; ++j) {
                const int ridx = mi * 8 + j;
                const int qa = q0 + mi * 16 + hi * 8 + j;
                const int ka0 = kc0 + ln;
                const int ka1 = kc0 + 16 + ln;
                float s0 = sacc[mi][0][j] * 0.125f;
                float s1 = sacc[mi][1][j] * 0.125f;
                const int d0 = ka0 - qa, d1 = ka1 - qa;
                const bool ok0 = (ka0 >= 0) && (ka0 < Sq) && (d0 <= WINc) && (d0 >= -WINc);
                const bool ok1 = (ka1 >= 0) && (ka1 < Sq) && (d1 <= WINc) && (d1 >= -WINc);
                s0 = ok0 ? s0 : -1e9f;
                s1 = ok1 ? s1 : -1e9f;
                float lm = fmaxf(s0, s1);
                #pragma unroll
                for (int o = 8; o; o >>= 1) lm = fmaxf(lm, __shfl_xor(lm, o));
                const float mo = mrow[ridx];
                const float mn = fmaxf(mo, lm);
                const float alpha = __expf(mo - mn);
                const float p0 = __expf(s0 - mn);
                const float p1 = __expf(s1 - mn);
                float rs = p0 + p1;
                #pragma unroll
                for (int o = 8; o; o >>= 1) rs += __shfl_xor(rs, o);
                mrow[ridx] = mn;
                lrow[ridx] = lrow[ridx] * alpha + rs;
                #pragma unroll
                for (int ni = 0; ni < 4; ++ni) oacc[mi][ni][j] *= alpha;
                const int rloc = mi * 16 + hi * 8 + j;
                sP[wid][rloc][ln]      = (__bf16)p0;
                sP[wid][rloc][16 + ln] = (__bf16)p1;
            }
        }
        __syncthreads();

        // ---- O += P V ----
        FragBf vf[4];
        #pragma unroll
        for (int ni = 0; ni < 4; ++ni) {
            const int dl = ni * 16 + ln;
            vf[ni].q[0] = *(const uint4*)&sVT[dl][hi * 8];
            vf[ni].q[1] = *(const uint4*)&sVT[dl][hi * 8 + 16];
        }
        FragBf pf[2];
        #pragma unroll
        for (int mi = 0; mi < 2; ++mi) {
            const int pr = mi * 16 + ln;
            pf[mi].q[0] = *(const uint4*)&sP[wid][pr][hi * 8];
            pf[mi].q[1] = *(const uint4*)&sP[wid][pr][hi * 8 + 16];
        }
        #pragma unroll
        for (int mi = 0; mi < 2; ++mi)
            #pragma unroll
            for (int ni = 0; ni < 4; ++ni)
                oacc[mi][ni] = wmma_bf16(pf[mi], vf[ni], oacc[mi][ni]);
        __syncthreads();
    }

    // ---- normalize and write (bf16) ----
    #pragma unroll
    for (int mi = 0; mi < 2; ++mi)
        #pragma unroll
        for (int j = 0; j < 8; ++j) {
            const float inv = 1.0f / lrow[mi * 8 + j];
            const int qa = q0 + mi * 16 + hi * 8 + j;
            #pragma unroll
            for (int ni = 0; ni < 4; ++ni) {
                const int dc = ni * 16 + ln;
                Cp[(size_t)qa * HIDc + dc] = (__bf16)(oacc[mi][ni][j] * inv);
            }
        }
}

// =====================================================================
// Top head: out[b] = h[b,0,:] . top_w[0:768] + fts[b]*top_w[768] + top_b
// grid = Bz, block = 256
// =====================================================================
__global__ void top_kernel(const __bf16* __restrict__ h, const float* __restrict__ fts,
                           const float* __restrict__ tw, const float* __restrict__ tb,
                           float* __restrict__ out)
{
    const int b = blockIdx.x;
    const int tid = threadIdx.x;
    const __bf16* row = h + (size_t)b * Sq * HIDc;   // token 0 of batch b
    float p = (float)row[tid] * tw[tid] + (float)row[tid + 256] * tw[tid + 256]
            + (float)row[tid + 512] * tw[tid + 512];
    #pragma unroll
    for (int o = 16; o; o >>= 1) p += __shfl_xor(p, o);
    __shared__ float rp[8];
    if ((tid & 31) == 0) rp[tid >> 5] = p;
    __syncthreads();
    if (tid == 0) {
        float s = 0.0f;
        #pragma unroll
        for (int i = 0; i < 8; ++i) s += rp[i];
        out[b] = s + fts[b] * tw[HIDc] + tb[0];
    }
}

// =====================================================================
extern "C" void kernel_launch(void* const* d_in, const int* in_sizes, int n_in,
                              void* d_out, int out_size, void* d_ws, size_t ws_size,
                              hipStream_t stream)
{
    (void)in_sizes; (void)n_in; (void)out_size; (void)ws_size;

    const int*   ids  = (const int*)  d_in[0];
    // d_in[1] = mask (all ones; band/bounds masking handled in-kernel)
    const float* fts  = (const float*)d_in[2];
    const float* wemb = (const float*)d_in[3];
    const float* pemb = (const float*)d_in[4];
    const float* temb = (const float*)d_in[5];
    const float* eg   = (const float*)d_in[6];
    const float* ebb  = (const float*)d_in[7];
    const float* Wq   = (const float*)d_in[8];
    const float* Bq   = (const float*)d_in[9];
    const float* Wk   = (const float*)d_in[10];
    const float* Bk   = (const float*)d_in[11];
    const float* Wv   = (const float*)d_in[12];
    const float* Bv   = (const float*)d_in[13];
    const float* Wo   = (const float*)d_in[14];
    const float* Bo   = (const float*)d_in[15];
    const float* G1   = (const float*)d_in[16];
    const float* B1   = (const float*)d_in[17];
    const float* Wi   = (const float*)d_in[18];
    const float* Bi   = (const float*)d_in[19];
    const float* Wo2  = (const float*)d_in[20];
    const float* Bo2  = (const float*)d_in[21];
    const float* G2   = (const float*)d_in[22];
    const float* B2   = (const float*)d_in[23];
    const float* topw = (const float*)d_in[24];
    const float* topb = (const float*)d_in[25];

    __bf16* wsb = (__bf16*)d_ws;
    const size_t msz  = (size_t)Mrows * HIDc;
    const size_t fsz  = (size_t)Mrows * FFd;
    const size_t whh  = (size_t)Ll * HIDc * HIDc;
    const size_t whf  = (size_t)Ll * HIDc * FFd;

    __bf16* h    = wsb;
    __bf16* q    = h    + msz;
    __bf16* k    = q    + msz;
    __bf16* v    = k    + msz;
    __bf16* ctx  = v    + msz;
    __bf16* h1   = ctx  + msz;
    __bf16* ff   = h1   + msz;
    __bf16* WqT  = ff   + fsz;     // [L][768][768]  (bf16, [N][K])
    __bf16* WkT  = WqT  + whh;
    __bf16* WvT  = WkT  + whh;
    __bf16* WoT  = WvT  + whh;
    __bf16* WiT  = WoT  + whh;     // [L][3072][768]
    __bf16* Wo2T = WiT  + whf;     // [L][768][3072]

    // ---- one-time (per launch) weight convert + transpose to bf16 [N,K] ----
    convt_kernel<<<dim3(HIDc/64, HIDc/64, Ll), 256, 0, stream>>>(Wq,  WqT,  HIDc, HIDc);
    convt_kernel<<<dim3(HIDc/64, HIDc/64, Ll), 256, 0, stream>>>(Wk,  WkT,  HIDc, HIDc);
    convt_kernel<<<dim3(HIDc/64, HIDc/64, Ll), 256, 0, stream>>>(Wv,  WvT,  HIDc, HIDc);
    convt_kernel<<<dim3(HIDc/64, HIDc/64, Ll), 256, 0, stream>>>(Wo,  WoT,  HIDc, HIDc);
    convt_kernel<<<dim3(FFd/64,  HIDc/64, Ll), 256, 0, stream>>>(Wi,  WiT,  HIDc, FFd);
    convt_kernel<<<dim3(HIDc/64, FFd/64,  Ll), 256, 0, stream>>>(Wo2, Wo2T, FFd,  HIDc);

    embed_ln_kernel<<<Mrows, 256, 0, stream>>>(ids, wemb, pemb, temb, eg, ebb, h);

    const dim3 g768 (HIDc / 128, Mrows / 128);   // 6 x 64
    const dim3 g3072(FFd  / 128, Mrows / 128);   // 24 x 64
    const dim3 gattn(Sq / BLKc, Hh, Bz);         // 16 x 12 x 2

    for (int l = 0; l < Ll; ++l) {
        const size_t oHH = (size_t)l * HIDc * HIDc;
        const size_t oHF = (size_t)l * HIDc * FFd;
        const __bf16* wq = WqT  + oHH;  const float* bq = Bq + (size_t)l * HIDc;
        const __bf16* wk = WkT  + oHH;  const float* bk = Bk + (size_t)l * HIDc;
        const __bf16* wv = WvT  + oHH;  const float* bv = Bv + (size_t)l * HIDc;
        const __bf16* wo = WoT  + oHH;  const float* bo = Bo + (size_t)l * HIDc;
        const __bf16* wi = WiT  + oHF;  const float* bi = Bi + (size_t)l * FFd;
        const __bf16* w2 = Wo2T + oHF;  const float* b2 = Bo2 + (size_t)l * HIDc;
        const float* g1 = G1 + (size_t)l * HIDc; const float* bb1 = B1 + (size_t)l * HIDc;
        const float* g2 = G2 + (size_t)l * HIDc; const float* bb2 = B2 + (size_t)l * HIDc;

        gemm_kernel<0><<<g768, 256, 0, stream>>>(h, wq, bq, nullptr, q, Mrows, HIDc, HIDc);
        gemm_kernel<0><<<g768, 256, 0, stream>>>(h, wk, bk, nullptr, k, Mrows, HIDc, HIDc);
        gemm_kernel<0><<<g768, 256, 0, stream>>>(h, wv, bv, nullptr, v, Mrows, HIDc, HIDc);

        attn_kernel<<<gattn, 256, 0, stream>>>(q, k, v, ctx);

        gemm_kernel<2><<<g768, 256, 0, stream>>>(ctx, wo, bo, h, h1, Mrows, HIDc, HIDc);
        ln_kernel<<<Mrows, 256, 0, stream>>>(h1, g1, bb1);

        gemm_kernel<1><<<g3072, 256, 0, stream>>>(h1, wi, bi, nullptr, ff, Mrows, FFd, HIDc);
        gemm_kernel<2><<<g768, 256, 0, stream>>>(ff, w2, b2, h1, h, Mrows, HIDc, FFd);
        ln_kernel<<<Mrows, 256, 0, stream>>>(h, g2, bb2);
    }

    top_kernel<<<Bz, 256, 0, stream>>>(h, fts, topw, topb, (float*)d_out);
}
